// QuantizedLinear_11914239279226
// MI455X (gfx1250) — matrix operand order repaired and checked by reading, not compiled, not run on hardware
//
#include <hip/hip_runtime.h>

// QuantizedLinear: out[M,N] = (x[M,K] . Wq[N,K]^T) * scale[N] + bias[N]
// M=B*S=4096, N=OUT=4096, K=IN=4096. f16 WMMA (int8 weights are exact in f16),
// f32 accumulation, 128x128 workgroup tile, TK=64 double-buffered LDS.

#define TM 128
#define TN 128
#define TK 64
#define LDSTR 72  // halves per LDS row: 144B stride -> conflict-free b128 frag reads

typedef __attribute__((ext_vector_type(16))) _Float16 v16h;
typedef __attribute__((ext_vector_type(8)))  _Float16 v8h;
typedef __attribute__((ext_vector_type(8)))  float    v8f;
typedef __attribute__((ext_vector_type(4)))  float    f4;
typedef __attribute__((ext_vector_type(4)))  int      i4;

union frag16 { v16h v; v8h h[2]; };

__global__ __launch_bounds__(256)
void qlinear_wmma_f16(const float* __restrict__ X,
                      const int*   __restrict__ Wq,
                      const float* __restrict__ scale,
                      const float* __restrict__ bias,
                      float* __restrict__ Out,
                      int M, int N, int K)
{
    extern __shared__ _Float16 smem[];
    _Float16* As = smem;                      // [2][TM][LDSTR]
    _Float16* Bs = smem + 2 * TM * LDSTR;     // [2][TN][LDSTR]

    const int tid  = threadIdx.x;
    const int lane = tid & 31;
    const int wave = tid >> 5;

    const int bm = blockIdx.y * TM;
    const int bn = blockIdx.x * TN;

    // 2 (M) x 4 (N) wave grid; each wave computes a 64x32 output tile
    const int wm = (wave >> 2) * 64;
    const int wn = (wave & 3)  * 32;

    const int r  = lane & 15;   // row within 16x16 fragment
    const int kg = lane >> 4;   // half-wave group

    // Staging: each thread owns 32 consecutive source elements of one tile row
    const int srow = tid >> 1;            // 0..127
    const int scol = (tid & 1) * 32;      // 0 or 32

    const float* gA = X  + (size_t)(bm + srow) * K + scol;
    const int*   gB = Wq + (size_t)(bn + srow) * K + scol;

    v8f acc[4][2];
#pragma unroll
    for (int i = 0; i < 4; ++i)
#pragma unroll
        for (int j = 0; j < 2; ++j) acc[i][j] = (v8f)0.0f;

    // ---- global -> LDS staging with on-the-fly conversion to f16 ----
    auto stage = [&](int buf, int kt) {
        _Float16* a_dst = As + buf * (TM * LDSTR) + srow * LDSTR + scol;
        _Float16* b_dst = Bs + buf * (TM * LDSTR) + srow * LDSTR + scol;
        const float* ga = gA + kt;
        const int*   gb = gB + kt;
        if (kt + 2 * TK < K) {                    // speculative prefetch 2 tiles ahead
            __builtin_prefetch(ga + 2 * TK, 0, 0);
            __builtin_prefetch(gb + 2 * TK, 0, 0);
        }
#pragma unroll
        for (int v = 0; v < 4; ++v) {
            f4 x0 = *(const f4*)(ga + v * 8);
            f4 x1 = *(const f4*)(ga + v * 8 + 4);
            v8h ha;
            ha[0] = (_Float16)x0.x; ha[1] = (_Float16)x0.y;
            ha[2] = (_Float16)x0.z; ha[3] = (_Float16)x0.w;
            ha[4] = (_Float16)x1.x; ha[5] = (_Float16)x1.y;
            ha[6] = (_Float16)x1.z; ha[7] = (_Float16)x1.w;
            *(v8h*)(a_dst + v * 8) = ha;

            i4 w0 = *(const i4*)(gb + v * 8);
            i4 w1 = *(const i4*)(gb + v * 8 + 4);
            v8h hb;
            hb[0] = (_Float16)w0.x; hb[1] = (_Float16)w0.y;
            hb[2] = (_Float16)w0.z; hb[3] = (_Float16)w0.w;
            hb[4] = (_Float16)w1.x; hb[5] = (_Float16)w1.y;
            hb[6] = (_Float16)w1.z; hb[7] = (_Float16)w1.w;
            *(v8h*)(b_dst + v * 8) = hb;
        }
    };

    // ---- LDS -> fragments -> WMMA ----
    auto compute = [&](int buf) {
        const _Float16* a_base = As + buf * (TM * LDSTR);
        const _Float16* b_base = Bs + buf * (TM * LDSTR);
#pragma unroll
        for (int ks = 0; ks < TK; ks += 32) {
            v16h afrag[4], bfrag[2];
#pragma unroll
            for (int i = 0; i < 4; ++i) {
                // A 16x32 f16 layout: lane=M row; halves 0-15 = K kg*16..kg*16+15
                // (contiguous, matching B)
                const _Float16* p = a_base + (wm + i * 16 + r) * LDSTR + ks;
                frag16 u;
                u.h[0] = *(const v8h*)(p + kg * 16);
                u.h[1] = *(const v8h*)(p + kg * 16 + 8);
                afrag[i] = u.v;
            }
#pragma unroll
            for (int j = 0; j < 2; ++j) {
                // B 32x16 f16 layout: lane=N col; lanes 0-15 hold K 0-15,
                // lanes 16-31 hold K 16-31 (16 contiguous halves per lane)
                const _Float16* p = b_base + (wn + j * 16 + r) * LDSTR + ks + kg * 16;
                frag16 u;
                u.h[0] = *(const v8h*)(p);
                u.h[1] = *(const v8h*)(p + 8);
                bfrag[j] = u.v;
            }
#pragma unroll
            for (int i = 0; i < 4; ++i)
#pragma unroll
                for (int j = 0; j < 2; ++j)
                    acc[i][j] = __builtin_amdgcn_wmma_f32_16x16x32_f16(
                        /*neg_a=*/false, afrag[i],
                        /*neg_b=*/false, bfrag[j],
                        /*c_mod=*/(short)0, acc[i][j],
                        /*reuse_a=*/false, /*reuse_b=*/false);
        }
    };

    // ---- main K loop, double buffered ----
    stage(0, 0);
    __syncthreads();
    const int nk = K / TK;
    for (int t = 0; t < nk; ++t) {
        const int buf = t & 1;
        if (t + 1 < nk) stage(buf ^ 1, (t + 1) * TK);
        compute(buf);
        __syncthreads();
    }

    // ---- epilogue: scale per output channel + bias ----
#pragma unroll
    for (int j = 0; j < 2; ++j) {
        const int col = bn + wn + j * 16 + r;
        const float s = scale[col];
        const float b = bias[col];
#pragma unroll
        for (int i = 0; i < 4; ++i) {
            // C/D layout: VGPR e holds M = kg*8 + e within the 16x16 tile
            const int row0 = bm + wm + i * 16 + kg * 8;
            float* o = Out + (size_t)row0 * N + col;
#pragma unroll
            for (int e = 0; e < 8; ++e)
                o[(size_t)e * N] = acc[i][j][e] * s + b;
        }
    }
}

extern "C" void kernel_launch(void* const* d_in, const int* in_sizes, int n_in,
                              void* d_out, int out_size, void* d_ws, size_t ws_size,
                              hipStream_t stream) {
    const float* x     = (const float*)d_in[0];   // [B,S,IN] fp32
    const int*   wq    = (const int*)d_in[1];     // [OUT,IN] int (quantized int8 values)
    const float* scale = (const float*)d_in[2];   // [1,OUT]
    const float* bias  = (const float*)d_in[3];   // [OUT]
    float*       out   = (float*)d_out;           // [B,S,OUT]

    const int N = in_sizes[3];            // OUT
    const int K = in_sizes[1] / N;        // IN
    const int M = in_sizes[0] / K;        // B*S

    dim3 grid(N / TN, M / TM);
    dim3 block(256);
    // 2 buffers x (A tile + B tile), each 128 x LDSTR halves
    size_t shmem = (size_t)2 * 2 * TM * LDSTR * sizeof(_Float16);  // 73,728 B
    qlinear_wmma_f16<<<grid, block, shmem, stream>>>(x, wq, scale, bias, out, M, N, K);
}
